// GlobalAttentionPool_53901839564896
// MI455X (gfx1250) — compile-verified, matrix operand order checked by reading, not run on hardware
//
#include <hip/hip_runtime.h>

typedef __attribute__((ext_vector_type(16))) _Float16 v16h;
typedef __attribute__((ext_vector_type(8)))  float    v8f;
typedef __attribute__((ext_vector_type(4)))  float    v4f;

#define FEAT 128  // D

// ---------------------------------------------------------------------------
// Zero-fill the [N, 128] output (harness poisons d_out; untouched rows must
// be exactly zero). float4 stores, fully coalesced.
// ---------------------------------------------------------------------------
__global__ void gap_zero_kernel(float* __restrict__ out, long long n4) {
    long long i = (long long)blockIdx.x * blockDim.x + threadIdx.x;
    if (i < n4) {
        v4f z = {0.f, 0.f, 0.f, 0.f};
        ((v4f*)out)[i] = z;
    }
}

__device__ __forceinline__ float bcast_lane(float v, int srcLane) {
    return __builtin_bit_cast(float,
        __builtin_amdgcn_readlane(__builtin_bit_cast(int, v), srcLane));
}

// ---------------------------------------------------------------------------
// One wave (32 lanes) handles 16 rows of x.
// Phase 1: gate logits for the 16 rows via v_wmma_f32_16x16x32_f16 (K=128 as
//          4 chained WMMAs). B = W_gate chunk replicated across all 16
//          columns, so lane l holds the logits of rows 8*(l>>4) + 0..7.
//          Out-of-range rows are CLAMPED (their logits are never consumed).
// Phase 2: each lane owns 4 contiguous features; walk the 16 (sorted-batch)
//          rows accumulating gate*x in registers per segment, flush segment
//          sums with global_atomic_add_f32.
// ---------------------------------------------------------------------------
__global__ __launch_bounds__(256) void gap_scatter_kernel(
    const float* __restrict__ x,
    const int*   __restrict__ batch,
    const float* __restrict__ Wg,
    const float* __restrict__ bg,
    float*       __restrict__ out,
    int N)
{
    const int lane        = threadIdx.x & 31;
    const int waveInBlock = threadIdx.x >> 5;
    const long long wave  = (long long)blockIdx.x * 8 + waveInBlock;
    const long long row0  = wave * 16;
    if (row0 >= (long long)N) return;   // wave-uniform exit

    // ---- Phase 1: WMMA gate logits -------------------------------------
    // A layout (16-bit 16x32): lane l -> row M = l&15; half = l>>4.
    //   v16h elems 0..7  = K (k0 + 8*half + 0..7)
    //   v16h elems 8..15 = K (k0 + 16 + 8*half + 0..7)
    const int m  = lane & 15;
    const int hi = lane >> 4;
    long long rowA = row0 + m;
    if (rowA >= (long long)N) rowA = (long long)N - 1;   // clamp: unconditional loads
    const float* xrow = x + rowA * FEAT;
    const int kbase = 8 * hi;

    v8f c = {0.f, 0.f, 0.f, 0.f, 0.f, 0.f, 0.f, 0.f};
#pragma unroll
    for (int kc = 0; kc < 4; ++kc) {
        const int kl = kc * 32 + kbase;        // low 16-K group start
        // vectorized b128 loads: 8 floats per K-group
        v4f xa0 = *(const v4f*)(xrow + kl);
        v4f xa1 = *(const v4f*)(xrow + kl + 4);
        v4f xa2 = *(const v4f*)(xrow + kl + 16);
        v4f xa3 = *(const v4f*)(xrow + kl + 20);
        v4f wb0 = *(const v4f*)(Wg + kl);
        v4f wb1 = *(const v4f*)(Wg + kl + 4);
        v4f wb2 = *(const v4f*)(Wg + kl + 16);
        v4f wb3 = *(const v4f*)(Wg + kl + 20);

        v16h a, b;
#pragma unroll
        for (int j = 0; j < 4; ++j) {
            a[j]      = (_Float16)xa0[j];
            a[j + 4]  = (_Float16)xa1[j];
            a[j + 8]  = (_Float16)xa2[j];
            a[j + 12] = (_Float16)xa3[j];
            b[j]      = (_Float16)wb0[j];
            b[j + 4]  = (_Float16)wb1[j];
            b[j + 8]  = (_Float16)wb2[j];
            b[j + 12] = (_Float16)wb3[j];
        }
        c = __builtin_amdgcn_wmma_f32_16x16x32_f16(
                /*neg_a=*/false, a, /*neg_b=*/false, b,
                /*c_mod=*/(short)0, c, /*reuse_a=*/false, /*reuse_b=*/false);
    }

    // c[v] on lane l = logit of row M = v + 8*(l>>4)  (duplicated over N cols)
    const float bias = bg[0];
    float g[8];
#pragma unroll
    for (int i = 0; i < 8; ++i) {
        float z = c[i] + bias;
        g[i] = 1.0f / (1.0f + __expf(-z));
    }

    // ---- Phase 2: segment-reduced scatter ------------------------------
    const int col = lane * 4;        // 4 contiguous floats per lane
    v4f acc = {0.f, 0.f, 0.f, 0.f};
    int cur = -1;                    // wave-uniform (batch[] uniform per row)
#pragma unroll
    for (int r = 0; r < 16; ++r) {
        long long row = row0 + r;
        const bool ok = row < (long long)N;          // wave-uniform
        if (!ok) row = (long long)N - 1;             // clamp, contribution zeroed
        const int bi = batch[row];
        // broadcast gate(r): rows 0..7 live on lane 0, rows 8..15 on lane 16
        float gr = (r < 8) ? bcast_lane(g[r], 0) : bcast_lane(g[r - 8], 16);
        if (!ok) gr = 0.f;

        if (bi != cur) {
            if (cur >= 0) {
                float* o = out + (long long)cur * FEAT + col;
                unsafeAtomicAdd(o + 0, acc.x);
                unsafeAtomicAdd(o + 1, acc.y);
                unsafeAtomicAdd(o + 2, acc.z);
                unsafeAtomicAdd(o + 3, acc.w);
            }
            acc = (v4f){0.f, 0.f, 0.f, 0.f};
            cur = bi;
        }
        v4f xv = *(const v4f*)(x + row * FEAT + col);
        acc.x += xv.x * gr;
        acc.y += xv.y * gr;
        acc.z += xv.z * gr;
        acc.w += xv.w * gr;
    }
    if (cur >= 0) {
        float* o = out + (long long)cur * FEAT + col;
        unsafeAtomicAdd(o + 0, acc.x);
        unsafeAtomicAdd(o + 1, acc.y);
        unsafeAtomicAdd(o + 2, acc.z);
        unsafeAtomicAdd(o + 3, acc.w);
    }
}

// ---------------------------------------------------------------------------
extern "C" void kernel_launch(void* const* d_in, const int* in_sizes, int n_in,
                              void* d_out, int out_size, void* d_ws, size_t ws_size,
                              hipStream_t stream) {
    const float* x     = (const float*)d_in[0];   // [N, 128] f32
    const int*   batch = (const int*)d_in[1];     // [N] sorted graph ids
    const float* Wg    = (const float*)d_in[2];   // [128, 1] f32
    const float* bg    = (const float*)d_in[3];   // [1] f32
    float*       out   = (float*)d_out;           // [N, 128] f32

    const int N = in_sizes[1];                    // node count from batch[]

    // 1) zero the whole output
    const long long n4 = (long long)out_size / 4;
    const int zblocks  = (int)((n4 + 255) / 256);
    gap_zero_kernel<<<zblocks, 256, 0, stream>>>(out, n4);

    // 2) gated segment-sum scatter (8 waves / block, 16 rows / wave)
    const long long waves  = ((long long)N + 15) / 16;
    const int       blocks = (int)((waves + 7) / 8);
    gap_scatter_kernel<<<blocks, 256, 0, stream>>>(x, batch, Wg, bg, out, N);
}
